// RWKV6Attention_50964081934479
// MI455X (gfx1250) — matrix operand order repaired and checked by reading, not compiled, not run on hardware
//
#include <hip/hip_runtime.h>
#include <hip/hip_bf16.h>

typedef __attribute__((ext_vector_type(16))) __bf16 v16bf;
typedef __attribute__((ext_vector_type(8)))  float  v8f;

#define D_MODEL 1024
#define SEQ_T   2048
#define BATCH   2
#define M_TOK   (BATCH*SEQ_T)
#define NHEAD   16
#define HDIM    64
#define NMIXCOL 160   // 5 * 32
#define DMIXR   32
#define DDEC    64

static __device__ __forceinline__ float bf2f(__bf16 b) {
  unsigned short us = __builtin_bit_cast(unsigned short, b);
  return __uint_as_float(((unsigned)us) << 16);
}
static __device__ __forceinline__ __bf16 f2bf(float f) {
  unsigned u = __float_as_uint(f);
  unsigned r = (u + 0x7FFFu + ((u >> 16) & 1u)) >> 16;  // round-to-nearest-even
  return __builtin_bit_cast(__bf16, (unsigned short)r);
}

// ---------------------------------------------------------------------------
// Transpose + fp32 -> bf16 convert: W (R x C, row major) -> Wt (C x R)
// ---------------------------------------------------------------------------
__global__ void rwkv_transpose_bf16(const float* __restrict__ W,
                                    __bf16* __restrict__ Wt, int R, int C) {
  long id = (long)blockIdx.x * blockDim.x + threadIdx.x;
  if (id >= (long)R * C) return;
  int r = (int)(id / C);
  int c = (int)(id % C);
  Wt[(size_t)c * R + r] = f2bf(W[id]);
}

// ---------------------------------------------------------------------------
// Token shift: dxprev = x_{t-1} - x_t ; xxx = x + dxprev * time_maa_x (bf16)
// ---------------------------------------------------------------------------
__global__ void rwkv_prep(const float* __restrict__ x, const float* __restrict__ tmx,
                          float* __restrict__ dxprev, __bf16* __restrict__ xxx) {
  long id = (long)blockIdx.x * blockDim.x + threadIdx.x;
  if (id >= (long)M_TOK * D_MODEL) return;
  int d = (int)(id % D_MODEL);
  long m = id / D_MODEL;
  int tloc = (int)(m % SEQ_T);
  float xc = x[id];
  float xp = (tloc == 0) ? 0.f : x[id - D_MODEL];
  float dx = xp - xc;
  dxprev[id] = dx;
  xxx[id] = f2bf(xc + dx * tmx[d]);
}

// ---------------------------------------------------------------------------
// xs_s = x + dxprev * (time_maa_s + mix_s)  for s in {r,k,v,w,g}   (bf16 out)
// ---------------------------------------------------------------------------
__global__ void rwkv_mix_apply(const float* __restrict__ x, const float* __restrict__ dxprev,
                               const float* __restrict__ tmr, const float* __restrict__ tmk,
                               const float* __restrict__ tmv, const float* __restrict__ tmw,
                               const float* __restrict__ tmg,
                               const __bf16* __restrict__ mixs, __bf16* __restrict__ xs) {
  long id = (long)blockIdx.x * blockDim.x + threadIdx.x;
  const long MD = (long)M_TOK * D_MODEL;
  if (id >= MD) return;
  int d = (int)(id % D_MODEL);
  float xc = x[id], dx = dxprev[id];
  const float* tms[5] = {tmr, tmk, tmv, tmw, tmg};
  #pragma unroll
  for (int s = 0; s < 5; ++s)
    xs[(long)s * MD + id] = f2bf(xc + dx * (tms[s][d] + bf2f(mixs[(long)s * MD + id])));
}

// ---------------------------------------------------------------------------
// og = bf16(o * gate)
// ---------------------------------------------------------------------------
__global__ void rwkv_gate_fuse(const float* __restrict__ o, const float* __restrict__ gate,
                               __bf16* __restrict__ og) {
  long id = (long)blockIdx.x * blockDim.x + threadIdx.x;
  if (id >= (long)M_TOK * D_MODEL) return;
  og[id] = f2bf(o[id] * gate[id]);
}

// ---------------------------------------------------------------------------
// LDS staging, OOB -> 0.  A tile: 128 rows x 64 cols (thread t: row t/2,
// 32-elem half t&1).  B tile: 256 rows x 64 cols (thread t: row t, both
// 32-elem halves).
// ---------------------------------------------------------------------------
static __device__ __forceinline__ void stage_seg32(const __bf16* __restrict__ src,
                                                   int ldsrc, int rows, int cols,
                                                   int gr, int gc, __bf16* d) {
  if (gr < rows && gc + 32 <= cols) {
    const uint4* s = reinterpret_cast<const uint4*>(src + (size_t)gr * ldsrc + gc);
    uint4* dd = reinterpret_cast<uint4*>(d);
    dd[0] = s[0]; dd[1] = s[1]; dd[2] = s[2]; dd[3] = s[3];
  } else {
    const __bf16 z = __builtin_bit_cast(__bf16, (unsigned short)0);
    for (int e = 0; e < 32; ++e) {
      int c = gc + e;
      d[e] = (gr < rows && c < cols) ? src[(size_t)gr * ldsrc + c] : z;
    }
  }
}

static __device__ __forceinline__ void stage_A(const __bf16* __restrict__ src,
                                               int ldsrc, int rows, int cols,
                                               int r0, int c0, __bf16* dst) {
  int t = threadIdx.x;
  int row = t >> 1, seg = t & 1;
  stage_seg32(src, ldsrc, rows, cols, r0 + row, c0 + seg * 32,
              dst + row * 64 + seg * 32);
}

static __device__ __forceinline__ void stage_B(const __bf16* __restrict__ src,
                                               int ldsrc, int rows, int cols,
                                               int r0, int c0, __bf16* dst) {
  int row = threadIdx.x;                    // 256 rows, 1 per thread
  #pragma unroll
  for (int seg = 0; seg < 2; ++seg)
    stage_seg32(src, ldsrc, rows, cols, r0 + row, c0 + seg * 32,
                dst + row * 64 + seg * 32);
}

union FragU { v16bf v; uint4 u[2]; };

// ---------------------------------------------------------------------------
// bf16 WMMA GEMM:  C(MxN) = op( A(MxK) * Bt(NxK)^T + bias )
//   op: 0 = identity, 1 = tanh, 2 = silu
//   output: Cb (bf16) if non-null, else Cf (fp32)
// Block tile 128(M) x 256(N), 8 waves (2x4), wave tile 64x64.
// Double-buffered LDS (96KB), one barrier per 64-wide K stage,
// 32 v_wmma per stage.  Fragment layouts per CDNA5 ISA 7.12.2.
// ---------------------------------------------------------------------------
__global__ __launch_bounds__(256)
void rwkv_wmma_gemm(const __bf16* __restrict__ A, int lda,
                    const __bf16* __restrict__ Bt, int ldb,
                    float* __restrict__ Cf, __bf16* __restrict__ Cb, int ldc,
                    const float* __restrict__ bias,
                    int M, int N, int K, int op) {
  __shared__ __align__(16) __bf16 As[2][128 * 64];
  __shared__ __align__(16) __bf16 Bs[2][256 * 64];

  const int m0 = blockIdx.y * 128;
  const int n0 = blockIdx.x * 256;
  const int t = threadIdx.x;
  const int w = t >> 5, lane = t & 31;
  const int wm = w >> 2, wn = w & 3;            // wave grid 2(M) x 4(N)
  const int lhalf = lane >> 4, lrow = lane & 15;

  v8f c[4][4];
  #pragma unroll
  for (int mi = 0; mi < 4; ++mi)
    #pragma unroll
    for (int ni = 0; ni < 4; ++ni)
      #pragma unroll
      for (int r = 0; r < 8; ++r) c[mi][ni][r] = 0.f;

  const int nst = (K + 63) / 64;
  stage_A(A,  lda, M, K, m0, 0, As[0]);
  stage_B(Bt, ldb, N, K, n0, 0, Bs[0]);

  for (int st = 0; st < nst; ++st) {
    __syncthreads();                       // buffer st&1 ready
    if (st + 1 < nst) {                    // overlap: stage next K-tile
      stage_A(A,  lda, M, K, m0, (st + 1) * 64, As[(st + 1) & 1]);
      stage_B(Bt, ldb, N, K, n0, (st + 1) * 64, Bs[(st + 1) & 1]);
    }
    if (st + 2 < nst) {                    // speculative prefetch (global_prefetch_b8)
      __builtin_prefetch(A  + (size_t)(m0 + (t >> 1)) * lda + (st + 2) * 64, 0, 1);
      __builtin_prefetch(Bt + (size_t)(n0 + t) * ldb + (st + 2) * 64, 0, 1);
    }
    const __bf16* Ab = As[st & 1];
    const __bf16* Bb = Bs[st & 1];

    #pragma unroll
    for (int ks = 0; ks < 2; ++ks) {
      FragU a[4];
      #pragma unroll
      for (int mi = 0; mi < 4; ++mi) {
        int row = wm * 64 + mi * 16 + lrow;
        const __bf16* p = Ab + row * 64 + ks * 32;
        a[mi].u[0] = *reinterpret_cast<const uint4*>(p + lhalf * 8);       // K = h*8 .. h*8+7
        a[mi].u[1] = *reinterpret_cast<const uint4*>(p + 16 + lhalf * 8);  // K = 16+h*8 ..
      }
      FragU bfr[4];
      #pragma unroll
      for (int ni = 0; ni < 4; ++ni) {
        int nr = wn * 64 + ni * 16 + lrow;
        const __bf16* p = Bb + nr * 64 + ks * 32;                          // K = h*8 .., 16+h*8 ..
        bfr[ni].u[0] = *reinterpret_cast<const uint4*>(p + lhalf * 8);
        bfr[ni].u[1] = *reinterpret_cast<const uint4*>(p + 16 + lhalf * 8);
      }
      #pragma unroll
      for (int mi = 0; mi < 4; ++mi)
        #pragma unroll
        for (int ni = 0; ni < 4; ++ni)
          c[mi][ni] = __builtin_amdgcn_wmma_f32_16x16x32_bf16(
              false, a[mi].v, false, bfr[ni].v, (short)0, c[mi][ni], false, false);
    }
  }

  // Epilogue: C layout lane 0-15 -> N=lane, M=r; lane 16-31 -> N=lane-16, M=8+r
  #pragma unroll
  for (int mi = 0; mi < 4; ++mi) {
    #pragma unroll
    for (int ni = 0; ni < 4; ++ni) {
      #pragma unroll
      for (int r = 0; r < 8; ++r) {
        int row = m0 + wm * 64 + mi * 16 + lhalf * 8 + r;
        int col = n0 + wn * 64 + ni * 16 + lrow;
        if (row < M && col < N) {
          float x = c[mi][ni][r];
          if (bias) x += bias[col];
          if (op == 1)      x = tanhf(x);
          else if (op == 2) x = x / (1.f + __expf(-x));   // silu
          if (Cb) Cb[(size_t)row * ldc + col] = f2bf(x);
          else    Cf[(size_t)row * ldc + col] = x;
        }
      }
    }
  }
}

// ---------------------------------------------------------------------------
// Gated linear attention scan. One block per (b,h). 256 threads.
// Thread t: v-column vi = t&63, k-range [kg*16, kg*16+16) with kg = t>>6.
// State S[64][64] lives in registers (16 floats/thread).
//   g = -exp(decay); eg = exp(g); k' = k*(1-eg); q' = q*HD^-0.5
//   S = S*eg[k] + k'[k]*v[v];  o[v] = sum_k q'[k]*S[k][v]
// ---------------------------------------------------------------------------
__global__ __launch_bounds__(256)
void rwkv_scan(const float* __restrict__ q, const float* __restrict__ k,
               const float* __restrict__ v, const float* __restrict__ decay,
               float* __restrict__ o) {
  int bh = blockIdx.x;
  int b = bh / NHEAD, h = bh % NHEAD;
  int t = threadIdx.x;
  int vi = t & 63, kg = t >> 6;

  __shared__ float qs[64], ks[64], vs[64], egs[64], opart[256];

  float s[16];
  #pragma unroll
  for (int j = 0; j < 16; ++j) s[j] = 0.f;

  const size_t base = (size_t)b * SEQ_T * D_MODEL + (size_t)h * HDIM;
  for (int tt = 0; tt < SEQ_T; ++tt) {
    size_t rowoff = base + (size_t)tt * D_MODEL;
    if (t < 64)        qs[t] = q[rowoff + t] * 0.125f;                       // HD^-0.5
    else if (t < 128)  { int i = t - 64;  egs[i] = __expf(-__expf(decay[rowoff + i])); }
    else if (t < 192)  { int i = t - 128; ks[i] = k[rowoff + i]; }
    else               { int i = t - 192; vs[i] = v[rowoff + i]; }
    __syncthreads();

    float acc = 0.f;
    float vv = vs[vi];
    #pragma unroll
    for (int j = 0; j < 16; ++j) {
      int kk = kg * 16 + j;
      float eg = egs[kk];
      float kval = ks[kk] * (1.f - eg);   // bonus: k * (1 - exp(g_log))
      s[j] = s[j] * eg + kval * vv;
      acc += qs[kk] * s[j];
    }
    opart[t] = acc;
    __syncthreads();
    if (t < 64)
      o[rowoff + t] = opart[t] + opart[64 + t] + opart[128 + t] + opart[192 + t];
    __syncthreads();
  }
}

// ---------------------------------------------------------------------------
extern "C" void kernel_launch(void* const* d_in, const int* in_sizes, int n_in,
                              void* d_out, int out_size, void* d_ws, size_t ws_size,
                              hipStream_t stream) {
  const float* x    = (const float*)d_in[0];
  const float* tmx  = (const float*)d_in[1];
  const float* tmr  = (const float*)d_in[2];
  const float* tmk  = (const float*)d_in[3];
  const float* tmv  = (const float*)d_in[4];
  const float* tmw  = (const float*)d_in[5];
  const float* tmg  = (const float*)d_in[6];
  const float* w1   = (const float*)d_in[7];   // (1024, 160)
  const float* w2   = (const float*)d_in[8];   // (5, 32, 1024)
  const float* tdec = (const float*)d_in[9];   // (1024,)
  const float* dw1  = (const float*)d_in[10];  // (1024, 64)
  const float* dw2  = (const float*)d_in[11];  // (64, 1024)
  const float* Wq   = (const float*)d_in[12];
  const float* Wk   = (const float*)d_in[13];
  const float* Wv   = (const float*)d_in[14];
  const float* Wo   = (const float*)d_in[15];
  const float* Wg   = (const float*)d_in[16];
  const float* bg   = (const float*)d_in[17];

  const long MD = (long)M_TOK * D_MODEL;

  char* ws = (char*)d_ws;
  size_t off = 0;
  auto alloc = [&](size_t bytes) -> void* {
    void* p = ws + off;
    off += (bytes + 255) & ~(size_t)255;
    return p;
  };
  float*  dxprev = (float*) alloc(MD * 4);
  __bf16* xxx    = (__bf16*)alloc(MD * 2);
  __bf16* w1t    = (__bf16*)alloc((size_t)NMIXCOL * D_MODEL * 2);
  __bf16* wqt    = (__bf16*)alloc((size_t)D_MODEL * D_MODEL * 2);
  __bf16* wkt    = (__bf16*)alloc((size_t)D_MODEL * D_MODEL * 2);
  __bf16* wvt    = (__bf16*)alloc((size_t)D_MODEL * D_MODEL * 2);
  __bf16* wgt    = (__bf16*)alloc((size_t)D_MODEL * D_MODEL * 2);
  __bf16* wot    = (__bf16*)alloc((size_t)D_MODEL * D_MODEL * 2);
  __bf16* dw1t   = (__bf16*)alloc((size_t)DDEC * D_MODEL * 2);
  __bf16* dw2t   = (__bf16*)alloc((size_t)D_MODEL * DDEC * 2);
  __bf16* w2t    = (__bf16*)alloc((size_t)5 * D_MODEL * DMIXR * 2);
  __bf16* mixpre = (__bf16*)alloc((size_t)M_TOK * NMIXCOL * 2);
  __bf16* mixs   = (__bf16*)alloc((size_t)5 * MD * 2);
  __bf16* xs     = (__bf16*)alloc((size_t)5 * MD * 2);
  float*  qb     = (float*) alloc(MD * 4);
  float*  kb     = (float*) alloc(MD * 4);
  float*  vb     = (float*) alloc(MD * 4);
  float*  gateb  = (float*) alloc(MD * 4);
  float*  decayb = (float*) alloc(MD * 4);
  __bf16* dec1   = (__bf16*)alloc((size_t)M_TOK * DDEC * 2);
  float*  ob     = (float*) alloc(MD * 4);
  __bf16* og     = (__bf16*)alloc(MD * 2);
  (void)ws_size; (void)in_sizes; (void)n_in; (void)out_size;

  auto cdiv = [](long a, long b) { return (int)((a + b - 1) / b); };

  // 1) Weight convert + transpose -> bf16 (N x K layout)
  rwkv_transpose_bf16<<<cdiv((long)D_MODEL * NMIXCOL, 256), 256, 0, stream>>>(w1, w1t, D_MODEL, NMIXCOL);
  rwkv_transpose_bf16<<<cdiv((long)D_MODEL * D_MODEL, 256), 256, 0, stream>>>(Wq, wqt, D_MODEL, D_MODEL);
  rwkv_transpose_bf16<<<cdiv((long)D_MODEL * D_MODEL, 256), 256, 0, stream>>>(Wk, wkt, D_MODEL, D_MODEL);
  rwkv_transpose_bf16<<<cdiv((long)D_MODEL * D_MODEL, 256), 256, 0, stream>>>(Wv, wvt, D_MODEL, D_MODEL);
  rwkv_transpose_bf16<<<cdiv((long)D_MODEL * D_MODEL, 256), 256, 0, stream>>>(Wg, wgt, D_MODEL, D_MODEL);
  rwkv_transpose_bf16<<<cdiv((long)D_MODEL * D_MODEL, 256), 256, 0, stream>>>(Wo, wot, D_MODEL, D_MODEL);
  rwkv_transpose_bf16<<<cdiv((long)D_MODEL * DDEC, 256), 256, 0, stream>>>(dw1, dw1t, D_MODEL, DDEC);
  rwkv_transpose_bf16<<<cdiv((long)DDEC * D_MODEL, 256), 256, 0, stream>>>(dw2, dw2t, DDEC, D_MODEL);
  for (int s = 0; s < 5; ++s)
    rwkv_transpose_bf16<<<cdiv((long)DMIXR * D_MODEL, 256), 256, 0, stream>>>(
        w2 + (size_t)s * DMIXR * D_MODEL, w2t + (size_t)s * D_MODEL * DMIXR, DMIXR, D_MODEL);

  // 2) Token shift
  rwkv_prep<<<cdiv(MD, 256), 256, 0, stream>>>(x, tmx, dxprev, xxx);

  dim3 blk(256);
  auto gemm_grid = [&](int N, int M) { return dim3((N + 255) / 256, (M + 127) / 128); };

  // 3) mix_pre = tanh(xxx @ w1)  (M x 160, bf16)
  rwkv_wmma_gemm<<<gemm_grid(NMIXCOL, M_TOK), blk, 0, stream>>>(
      xxx, D_MODEL, w1t, D_MODEL, nullptr, mixpre, NMIXCOL, nullptr,
      M_TOK, NMIXCOL, D_MODEL, 1);

  // 4) mix_s = mix_pre[:, s*32:(s+1)*32] @ w2[s]   (M x 1024, bf16), K=32
  for (int s = 0; s < 5; ++s)
    rwkv_wmma_gemm<<<gemm_grid(D_MODEL, M_TOK), blk, 0, stream>>>(
        mixpre + s * DMIXR, NMIXCOL, w2t + (size_t)s * D_MODEL * DMIXR, DMIXR,
        nullptr, mixs + (size_t)s * MD, D_MODEL, nullptr,
        M_TOK, D_MODEL, DMIXR, 0);

  // 5) xr,xk,xv,xw,xg (bf16)
  rwkv_mix_apply<<<cdiv(MD, 256), 256, 0, stream>>>(x, dxprev, tmr, tmk, tmv, tmw, tmg, mixs, xs);

  // 6) projections + decay LoRA + gate
  rwkv_wmma_gemm<<<gemm_grid(D_MODEL, M_TOK), blk, 0, stream>>>(
      xs + 0 * MD, D_MODEL, wqt, D_MODEL, qb, nullptr, D_MODEL, nullptr,
      M_TOK, D_MODEL, D_MODEL, 0);
  rwkv_wmma_gemm<<<gemm_grid(D_MODEL, M_TOK), blk, 0, stream>>>(
      xs + 1 * MD, D_MODEL, wkt, D_MODEL, kb, nullptr, D_MODEL, nullptr,
      M_TOK, D_MODEL, D_MODEL, 0);
  rwkv_wmma_gemm<<<gemm_grid(D_MODEL, M_TOK), blk, 0, stream>>>(
      xs + 2 * MD, D_MODEL, wvt, D_MODEL, vb, nullptr, D_MODEL, nullptr,
      M_TOK, D_MODEL, D_MODEL, 0);
  rwkv_wmma_gemm<<<gemm_grid(DDEC, M_TOK), blk, 0, stream>>>(
      xs + 3 * MD, D_MODEL, dw1t, D_MODEL, nullptr, dec1, DDEC, nullptr,
      M_TOK, DDEC, D_MODEL, 1);                                   // tanh
  rwkv_wmma_gemm<<<gemm_grid(D_MODEL, M_TOK), blk, 0, stream>>>(
      xs + 4 * MD, D_MODEL, wgt, D_MODEL, gateb, nullptr, D_MODEL, bg,
      M_TOK, D_MODEL, D_MODEL, 2);                                // silu(x+bg)
  rwkv_wmma_gemm<<<gemm_grid(D_MODEL, M_TOK), blk, 0, stream>>>(
      dec1, DDEC, dw2t, DDEC, decayb, nullptr, D_MODEL, tdec,
      M_TOK, D_MODEL, DDEC, 0);                                   // + time_decay

  // 7) recurrent gated linear attention
  rwkv_scan<<<BATCH * NHEAD, 256, 0, stream>>>(qb, kb, vb, decayb, ob);

  // 8) gate fuse + output projection
  rwkv_gate_fuse<<<cdiv(MD, 256), 256, 0, stream>>>(ob, gateb, og);
  rwkv_wmma_gemm<<<gemm_grid(D_MODEL, M_TOK), blk, 0, stream>>>(
      og, D_MODEL, wot, D_MODEL, (float*)d_out, nullptr, D_MODEL, nullptr,
      M_TOK, D_MODEL, D_MODEL, 0);
}
